// AttentiveMessagePassingLayer_79834852098597
// MI455X (gfx1250) — matrix operand order repaired and checked by reading, not compiled, not run on hardware
//
#include <hip/hip_runtime.h>
#include <math.h>

#define D_MODEL 128
#define D_ATTN  32
#define EFEAT   4
#define N_SPE   50000
#define N_RXN   100000
#define NEDGE   500000L
#define LN_EPS  1e-5f
#define INV_SCALE 0.17677669529663687f   // 1/sqrt(D_ATTN)

typedef __attribute__((ext_vector_type(16))) _Float16 v16h;
typedef __attribute__((ext_vector_type(8)))  float    v8f;

#define WPB 8                 // waves per block (wave32)
#define EPB (WPB * 16)        // edges per block

// -------- dynamic LDS layout (bytes) --------
#define OFF_W1   0            // 4*8*32*16 halfs = 32 KB (W1 B-fragments, f16)
#define OFF_W2   32768        // 32 KB
#define OFF_WK   65536        // 4*2*32*16 halfs = 8 KB
#define OFF_STG  73728        // 8 waves * 16*128 halfs = 32 KB (per-wave staging)
#define OFF_B1   106496       // 128 f32
#define OFF_B2   107008       // 128 f32
#define OFF_BK   107520       // 32 f32
#define OFF_W1F  107648       // 4*128 f32 (feat rows of W1)
#define LDS_BYTES 109696

__device__ __forceinline__ v16h load_afrag(const _Float16* p) {
  // 16-bit A-matrix swizzle: lane holds row (lane&15); K offsets are two
  // contiguous 8-half runs: [8*hi .. 8*hi+7] and [8*hi+16 .. 8*hi+23]
  v16h a;
#pragma unroll
  for (int j = 0; j < 8; ++j) { a[j] = p[j]; a[j + 8] = p[16 + j]; }
  return a;
}

__device__ __forceinline__ v16h load_bfrag(const _Float16* p) {
  v16h b;
#pragma unroll
  for (int j = 0; j < 16; ++j) b[j] = p[j];
  return b;
}

template <int NT>
__device__ __forceinline__ void wmma_gemm_k128(const _Float16* stg, const _Float16* sW,
                                               int r, int hi, int lane, v8f* acc) {
#pragma unroll
  for (int c = 0; c < 4; ++c) {
    v16h a = load_afrag(stg + r * D_MODEL + 32 * c + 8 * hi);
#pragma unroll
    for (int n = 0; n < NT; ++n) {
      v16h b = load_bfrag(sW + ((c * NT + n) * 32 + lane) * 16);
      acc[n] = __builtin_amdgcn_wmma_f32_16x16x32_f16(false, a, false, b,
                                                      (short)0, acc[n], false, false);
    }
  }
}

// =====================================================================
// Edge kernel: per wave, 16 edges.
//   X  = gather(h_src) (16x128 f16 in LDS)
//   H1 = silu(X@W1h + feat@W1f + b1)      (WMMA + f32 epilogue)
//   M  = H1@W2 + b2  -> msgs (f32, global) (WMMA)
//   K  = M@Wk + bk   -> kvec (f32, global) (WMMA)
// =====================================================================
__global__ __launch_bounds__(256) void edge_mlp_wmma(
    const float* __restrict__ h_src, const int* __restrict__ src_idx,
    const float* __restrict__ feat,
    const float* __restrict__ W1, const float* __restrict__ b1,
    const float* __restrict__ W2, const float* __restrict__ b2,
    const float* __restrict__ Wk, const float* __restrict__ bk,
    float* __restrict__ msgs, float* __restrict__ kvec, long nE) {
  extern __shared__ char smem[];
  _Float16* sW1 = (_Float16*)(smem + OFF_W1);
  _Float16* sW2 = (_Float16*)(smem + OFF_W2);
  _Float16* sWk = (_Float16*)(smem + OFF_WK);
  _Float16* stgAll = (_Float16*)(smem + OFF_STG);
  float* sB1 = (float*)(smem + OFF_B1);
  float* sB2 = (float*)(smem + OFF_B2);
  float* sBk = (float*)(smem + OFF_BK);
  float* sW1f = (float*)(smem + OFF_W1F);

  const int tid = threadIdx.x;

  // ---- stage weight B-fragments (f32 -> f16) ----
  for (int idx = tid; idx < 4 * 8 * 32 * 16; idx += 256) {
    int j = idx & 15, ln = (idx >> 4) & 31, n = (idx >> 9) & 7, c = idx >> 12;
    int k = 32 * c + 8 * (ln >> 4) + (j & 7) + 16 * (j >> 3);
    int col = 16 * n + (ln & 15);
    sW1[idx] = (_Float16)W1[k * D_MODEL + col];
    sW2[idx] = (_Float16)W2[k * D_MODEL + col];
  }
  for (int idx = tid; idx < 4 * 2 * 32 * 16; idx += 256) {
    int j = idx & 15, ln = (idx >> 4) & 31, n = (idx >> 9) & 1, c = idx >> 10;
    int k = 32 * c + 8 * (ln >> 4) + (j & 7) + 16 * (j >> 3);
    int col = 16 * n + (ln & 15);
    sWk[idx] = (_Float16)Wk[k * D_ATTN + col];
  }
  for (int i = tid; i < D_MODEL; i += 256) { sB1[i] = b1[i]; sB2[i] = b2[i]; }
  for (int i = tid; i < D_ATTN; i += 256) sBk[i] = bk[i];
  for (int i = tid; i < EFEAT * D_MODEL; i += 256) {
    int f = i >> 7, n = i & 127;
    sW1f[i] = W1[(D_MODEL + f) * D_MODEL + n];
  }
  __syncthreads();

  const int wave = tid >> 5;
  const int lane = tid & 31;
  const int r = lane & 15;
  const int hi = lane >> 4;
  _Float16* stg = stgAll + wave * (16 * D_MODEL);
  long e0 = ((long)blockIdx.x * WPB + wave) * 16;
  if (e0 >= nE) return;

  // ---- stage X: gather 16 rows of h_src, coalesced float4, cvt f16 ----
#pragma unroll
  for (int i = 0; i < 16; ++i) {
    int idx = i * 32 + lane;        // float4 index within 16x128 tile
    int row = idx >> 5;
    int c4 = (idx & 31) << 2;
    long e = e0 + row; if (e >= nE) e = nE - 1;
    const float4 v = *(const float4*)(h_src + (long)src_idx[e] * D_MODEL + c4);
    _Float16* s = stg + row * D_MODEL + c4;
    s[0] = (_Float16)v.x; s[1] = (_Float16)v.y;
    s[2] = (_Float16)v.z; s[3] = (_Float16)v.w;
  }

  // ---- GEMM1: X @ W1h ----
  v8f acc[8];
#pragma unroll
  for (int n = 0; n < 8; ++n)
#pragma unroll
    for (int k = 0; k < 8; ++k) acc[n][k] = 0.0f;
  wmma_gemm_k128<8>(stg, sW1, r, hi, lane, acc);

  // ---- epilogue 1: + feat@W1f + b1, silu, back to LDS as f16 ----
#pragma unroll
  for (int i = 0; i < 8; ++i) {
    int M = i + 8 * hi;
    long e = e0 + M; if (e >= nE) e = nE - 1;
    float f0 = feat[e * EFEAT + 0], f1 = feat[e * EFEAT + 1];
    float f2 = feat[e * EFEAT + 2], f3 = feat[e * EFEAT + 3];
#pragma unroll
    for (int n = 0; n < 8; ++n) {
      int N = 16 * n + r;
      float x = acc[n][i] + sB1[N] + f0 * sW1f[N] + f1 * sW1f[128 + N] +
                f2 * sW1f[256 + N] + f3 * sW1f[384 + N];
      float s = x / (1.0f + __expf(-x));       // silu
      stg[M * D_MODEL + N] = (_Float16)s;
      acc[n][i] = 0.0f;
    }
  }

  // ---- GEMM2: H1 @ W2 ----
  wmma_gemm_k128<8>(stg, sW2, r, hi, lane, acc);

  // ---- epilogue 2: + b2 -> msgs (global f32) and LDS f16 for GEMM3 ----
#pragma unroll
  for (int i = 0; i < 8; ++i) {
    int M = i + 8 * hi;
    long e = e0 + M;
#pragma unroll
    for (int n = 0; n < 8; ++n) {
      int N = 16 * n + r;
      float x = acc[n][i] + sB2[N];
      if (e < nE) msgs[e * D_MODEL + N] = x;
      stg[M * D_MODEL + N] = (_Float16)x;
    }
  }

  // ---- GEMM3: M @ Wk -> kvec ----
  v8f ak[2];
#pragma unroll
  for (int n = 0; n < 2; ++n)
#pragma unroll
    for (int k = 0; k < 8; ++k) ak[n][k] = 0.0f;
  wmma_gemm_k128<2>(stg, sWk, r, hi, lane, ak);

#pragma unroll
  for (int n = 0; n < 2; ++n)
#pragma unroll
    for (int i = 0; i < 8; ++i) {
      int M = i + 8 * hi;
      long e = e0 + M;
      int N = 16 * n + r;
      if (e < nE) kvec[e * D_ATTN + N] = ak[n][i] + sBk[N];
    }
}

// =====================================================================
// Small helper kernels
// =====================================================================
__global__ void zero_kernel(unsigned* __restrict__ p, long n) {
  long i = (long)blockIdx.x * blockDim.x + threadIdx.x;
  if (i < n) p[i] = 0u;
}

__global__ void qnode_kernel(const float* __restrict__ h, const float* __restrict__ Wq,
                             const float* __restrict__ bq, float* __restrict__ qn, int nn) {
  long t = (long)blockIdx.x * blockDim.x + threadIdx.x;
  if (t >= (long)nn * D_ATTN) return;
  long n = t >> 5;
  int d = (int)(t & 31);
  float acc = bq[d];
  for (int i = 0; i < D_MODEL; ++i) acc += h[n * D_MODEL + i] * Wq[i * D_ATTN + d];
  qn[t] = acc;
}

__device__ __forceinline__ unsigned encf(float f) {
  unsigned u = __float_as_uint(f);
  return (u & 0x80000000u) ? ~u : (u | 0x80000000u);
}
__device__ __forceinline__ float decf(unsigned u) {
  u = (u & 0x80000000u) ? (u & 0x7FFFFFFFu) : ~u;
  return __uint_as_float(u);
}

__global__ void raw_kernel(const float* __restrict__ qn, const float* __restrict__ kvec,
                           const int* __restrict__ dst_idx, float* __restrict__ raw,
                           unsigned* __restrict__ gmax, long nE) {
  long e = (long)blockIdx.x * blockDim.x + threadIdx.x;
  if (e >= nE) return;
  long dst = dst_idx[e];
  const float4* q4 = (const float4*)(qn + dst * D_ATTN);
  const float4* k4 = (const float4*)(kvec + e * D_ATTN);
  float s = 0.f;
#pragma unroll
  for (int i = 0; i < 8; ++i) {
    float4 a = q4[i], b = k4[i];
    s += a.x * b.x + a.y * b.y + a.z * b.z + a.w * b.w;
  }
  s *= INV_SCALE;
  raw[e] = s;
  atomicMax(gmax + dst, encf(s));
}

__global__ void expsum_kernel(const float* __restrict__ raw, const int* __restrict__ dst_idx,
                              const unsigned* __restrict__ gmax, float* __restrict__ wbuf,
                              float* __restrict__ gsum, long nE) {
  long e = (long)blockIdx.x * blockDim.x + threadIdx.x;
  if (e >= nE) return;
  long dst = dst_idx[e];
  float w = __expf(raw[e] - decf(gmax[dst]));
  wbuf[e] = w;
  atomicAdd(gsum + dst, w);
}

__global__ void scatter_kernel(const float* __restrict__ msgs, const float* __restrict__ wbuf,
                               const float* __restrict__ gsum, const int* __restrict__ dst_idx,
                               float* __restrict__ agg, long nE) {
  long t = (long)blockIdx.x * blockDim.x + threadIdx.x;
  if (t >= nE * D_MODEL) return;
  long e = t >> 7;
  int c = (int)(t & 127);
  long dst = dst_idx[e];
  float coef = wbuf[e] / (gsum[dst] + 1e-8f);
  atomicAdd(agg + dst * D_MODEL + c, msgs[t] * coef);
}

__global__ void ln_kernel(const float* __restrict__ h, const float* __restrict__ agg,
                          const float* __restrict__ g, const float* __restrict__ b,
                          float* __restrict__ out, int nn) {
  long t = (long)blockIdx.x * blockDim.x + threadIdx.x;
  long node = t >> 5;
  int lane = (int)(t & 31);
  if (node >= nn) return;
  float v[4];
  float sum = 0.f;
#pragma unroll
  for (int i = 0; i < 4; ++i) {
    int c = lane + 32 * i;
    v[i] = h[node * D_MODEL + c] + agg[node * D_MODEL + c];
    sum += v[i];
  }
#pragma unroll
  for (int off = 16; off > 0; off >>= 1) sum += __shfl_xor(sum, off, 32);
  float mu = sum * (1.f / 128.f);
  float var = 0.f;
#pragma unroll
  for (int i = 0; i < 4; ++i) { float d = v[i] - mu; var += d * d; }
#pragma unroll
  for (int off = 16; off > 0; off >>= 1) var += __shfl_xor(var, off, 32);
  var *= (1.f / 128.f);
  float inv = rsqrtf(var + LN_EPS);
#pragma unroll
  for (int i = 0; i < 4; ++i) {
    int c = lane + 32 * i;
    out[node * D_MODEL + c] = (v[i] - mu) * inv * g[c] + b[c];
  }
}

// =====================================================================
// Host-side pass driver
// =====================================================================
static void run_pass(const float* h_src, const float* h_dst,
                     const int* src_idx, const int* dst_idx, const float* feat,
                     const float* W1, const float* b1, const float* W2, const float* b2,
                     const float* Wq, const float* bq, const float* Wk, const float* bk,
                     const float* lng, const float* lnb, int n_dst, float* out,
                     float* msgs, float* kvec, float* qn, float* rawb, float* wbuf,
                     unsigned* gmax, float* gsum, float* agg, hipStream_t stream) {
  // gmax, gsum, agg are contiguous: one zeroing launch (n + n + 128n u32)
  long zn = (long)n_dst * 130;
  zero_kernel<<<(unsigned)((zn + 255) / 256), 256, 0, stream>>>(gmax, zn);

  qnode_kernel<<<(unsigned)(((long)n_dst * 32 + 255) / 256), 256, 0, stream>>>(
      h_dst, Wq, bq, qn, n_dst);

  unsigned nblk = (unsigned)((NEDGE + EPB - 1) / EPB);
  edge_mlp_wmma<<<nblk, 256, LDS_BYTES, stream>>>(h_src, src_idx, feat, W1, b1, W2, b2,
                                                  Wk, bk, msgs, kvec, NEDGE);

  raw_kernel<<<(unsigned)((NEDGE + 255) / 256), 256, 0, stream>>>(qn, kvec, dst_idx, rawb,
                                                                  gmax, NEDGE);
  expsum_kernel<<<(unsigned)((NEDGE + 255) / 256), 256, 0, stream>>>(rawb, dst_idx, gmax,
                                                                     wbuf, gsum, NEDGE);
  scatter_kernel<<<(unsigned)((NEDGE * D_MODEL + 255) / 256), 256, 0, stream>>>(
      msgs, wbuf, gsum, dst_idx, agg, NEDGE);
  ln_kernel<<<(unsigned)(((long)n_dst * 32 + 255) / 256), 256, 0, stream>>>(h_dst, agg, lng,
                                                                            lnb, out, n_dst);
}

extern "C" void kernel_launch(void* const* d_in, const int* in_sizes, int n_in,
                              void* d_out, int out_size, void* d_ws, size_t ws_size,
                              hipStream_t stream) {
  (void)in_sizes; (void)n_in; (void)out_size; (void)ws_size;
  const float* h_species   = (const float*)d_in[0];
  const float* h_reactions = (const float*)d_in[1];
  const int*   rs_rxn_idx  = (const int*)d_in[2];
  const int*   rs_spe_idx  = (const int*)d_in[3];
  const float* rs_feat     = (const float*)d_in[4];
  const int*   sr_spe_idx  = (const int*)d_in[5];
  const int*   sr_rxn_idx  = (const int*)d_in[6];
  const float* sr_feat     = (const float*)d_in[7];
  const float* W1_rs = (const float*)d_in[8];  const float* b1_rs = (const float*)d_in[9];
  const float* W2_rs = (const float*)d_in[10]; const float* b2_rs = (const float*)d_in[11];
  const float* Wq_rs = (const float*)d_in[12]; const float* bq_rs = (const float*)d_in[13];
  const float* Wk_rs = (const float*)d_in[14]; const float* bk_rs = (const float*)d_in[15];
  const float* ln_s_g = (const float*)d_in[16]; const float* ln_s_b = (const float*)d_in[17];
  const float* W1_sr = (const float*)d_in[18]; const float* b1_sr = (const float*)d_in[19];
  const float* W2_sr = (const float*)d_in[20]; const float* b2_sr = (const float*)d_in[21];
  const float* Wq_sr = (const float*)d_in[22]; const float* bq_sr = (const float*)d_in[23];
  const float* Wk_sr = (const float*)d_in[24]; const float* bk_sr = (const float*)d_in[25];
  const float* ln_r_g = (const float*)d_in[26]; const float* ln_r_b = (const float*)d_in[27];

  float* out_spe = (float*)d_out;
  float* out_rxn = out_spe + (long)N_SPE * D_MODEL;

  char* ws = (char*)d_ws;
  size_t ofs = 0;
  float* msgs = (float*)(ws + ofs); ofs += (size_t)NEDGE * D_MODEL * 4;
  float* kvec = (float*)(ws + ofs); ofs += (size_t)NEDGE * D_ATTN * 4;
  float* qn   = (float*)(ws + ofs); ofs += (size_t)N_RXN * D_ATTN * 4;
  float* rawb = (float*)(ws + ofs); ofs += (size_t)NEDGE * 4;
  float* wbuf = (float*)(ws + ofs); ofs += (size_t)NEDGE * 4;
  unsigned* gmax = (unsigned*)(ws + ofs); ofs += (size_t)N_RXN * 4;   // contiguous with
  float* gsum = (float*)(ws + ofs); ofs += (size_t)N_RXN * 4;        // gsum and agg for
  float* agg  = (float*)(ws + ofs); ofs += (size_t)N_RXN * D_MODEL * 4;  // single zeroing

  // Pass 1: reactions -> species
  run_pass(h_reactions, h_species, rs_rxn_idx, rs_spe_idx, rs_feat,
           W1_rs, b1_rs, W2_rs, b2_rs, Wq_rs, bq_rs, Wk_rs, bk_rs, ln_s_g, ln_s_b,
           N_SPE, out_spe, msgs, kvec, qn, rawb, wbuf, gmax, gsum, agg, stream);
  // Pass 2: (updated) species -> reactions
  run_pass(out_spe, h_reactions, sr_spe_idx, sr_rxn_idx, sr_feat,
           W1_sr, b1_sr, W2_sr, b2_sr, Wq_sr, bq_sr, Wk_sr, bk_sr, ln_r_g, ln_r_b,
           N_RXN, out_rxn, msgs, kvec, qn, rawb, wbuf, gmax, gsum, agg, stream);
}
